// DBSDTR_61701500175359
// MI455X (gfx1250) — compile-verified, matrix operand order checked by reading, not run on hardware
//
#include <hip/hip_runtime.h>
#include <hip/hip_bf16.h>
#include <math.h>

typedef __attribute__((ext_vector_type(16))) __bf16 bf16x16;
typedef __attribute__((ext_vector_type(8)))  __bf16 bf16x8;
typedef __attribute__((ext_vector_type(8)))  float  f32x8;

#define BATCH     16384
#define NTREES    64
#define TDEPTH    6
#define TREE_DIM  3
#define NCOLS     384          // NTREES * TDEPTH
#define NLAYERS   6
#define KSTRIDE   1056         // padded max fin (1050 -> 1056, multiple of 32)
#define SORT_N    2048

// ---------------------------------------------------------------------------
// Init: x (B x 90 f32) -> Xbf (B x KSTRIDE bf16); zero ALL per-layer K-pad
// columns (each layer pads fin -> fin+6; region is overwritten by that
// layer's own h outputs only AFTER its GEMM, so one-time zeroing is safe);
// zero d_out and reg accumulators.
// ---------------------------------------------------------------------------
__global__ __launch_bounds__(256) void k_init(const float* __restrict__ x,
                                              __bf16* __restrict__ Xbf,
                                              float* __restrict__ outv,
                                              float* __restrict__ regAcc) {
    int idx = blockIdx.x * 256 + threadIdx.x;
    const int perB = 96 + (NLAYERS - 1) * 6;   // 126
    int total = BATCH * perB;
    if (idx < total) {
        int b = idx / perB, f = idx % perB;
        if (f < 96) {
            float v = (f < 90) ? x[b * 90 + f] : 0.0f;
            Xbf[(size_t)b * KSTRIDE + f] = (__bf16)v;
        } else {
            int p = f - 96;                     // 0..29
            int layer = p / 6 + 1;              // layers 1..5
            int off   = p % 6;
            int col   = 90 + 192 * layer + off; // fin_layer .. fin_layer+5
            Xbf[(size_t)b * KSTRIDE + col] = (__bf16)0.0f;
        }
    }
    if (idx <= BATCH) outv[idx] = 0.0f;         // 16384 accumulators + reg slot
    if (idx < NLAYERS) regAcc[idx] = 0.0f;
}

// ---------------------------------------------------------------------------
// entmax15 over feature axis for one (tree,depth) column.
// fsl layout: [fin][NCOLS].  Output FW: [col][finpad] bf16 (pad K zeroed).
// ---------------------------------------------------------------------------
__global__ __launch_bounds__(256) void k_entmax(const float* __restrict__ fsl,
                                                __bf16* __restrict__ FW,
                                                int fin, int finpad) {
    __shared__ float z[KSTRIDE];
    __shared__ float s[SORT_N];
    __shared__ float red[256];
    __shared__ float ts;

    int col = blockIdx.x;
    int tid = threadIdx.x;

    for (int f = tid; f < fin; f += 256)
        z[f] = 0.5f * fsl[(size_t)f * NCOLS + col];
    __syncthreads();

    float m = -3.0e38f;
    for (int f = tid; f < fin; f += 256) m = fmaxf(m, z[f]);
    red[tid] = m;
    __syncthreads();
    for (int st = 128; st > 0; st >>= 1) {
        if (tid < st) red[tid] = fmaxf(red[tid], red[tid + st]);
        __syncthreads();
    }
    float mx = red[0];
    for (int f = tid; f < fin; f += 256) z[f] -= mx;
    __syncthreads();

    for (int i = tid; i < SORT_N; i += 256)
        s[i] = (i < fin) ? z[i] : -3.0e38f;
    __syncthreads();

    // bitonic sort, descending
    for (int ksz = 2; ksz <= SORT_N; ksz <<= 1) {
        for (int j = ksz >> 1; j > 0; j >>= 1) {
            for (int i = tid; i < SORT_N; i += 256) {
                int ixj = i ^ j;
                if (ixj > i) {
                    float a = s[i], b = s[ixj];
                    bool descBlk = ((i & ksz) == 0);
                    bool doswap = descBlk ? (a < b) : (a > b);
                    if (doswap) { s[i] = b; s[ixj] = a; }
                }
            }
            __syncthreads();
        }
    }

    if (tid == 0) {
        float cs = 0.f, cs2 = 0.f, tstar = s[0] - 1.0f;
        for (int i = 0; i < fin; ++i) {
            float v = s[i];
            cs += v; cs2 += v * v;
            float k = (float)(i + 1);
            float mean = cs / k;
            float msq  = cs2 / k;
            float delta = (1.f - k * (msq - mean * mean)) / k;
            float tau = mean - sqrtf(fmaxf(delta, 0.f));
            if (tau <= v) tstar = tau;
        }
        ts = tstar;
    }
    __syncthreads();
    float tstar = ts;

    for (int f = tid; f < finpad; f += 256) {
        float p = 0.f;
        if (f < fin) {
            float d = fmaxf(z[f] - tstar, 0.f);
            p = d * d;
        }
        FW[(size_t)col * finpad + f] = (__bf16)p;
    }
}

// ---------------------------------------------------------------------------
// GEMM (B x finpad) x (finpad x 384) bf16 WMMA, ping-pong LDS double buffer,
// entmoid gate epilogue.  Block tile 128x128, 8 waves (4M x 2N), each wave
// 2x4 frags of v_wmma_f32_16x16x32_bf16.
// ---------------------------------------------------------------------------
__device__ __forceinline__ float entmoid15_dev(float xg) {
    float aa  = fabsf(xg) * 0.5f;
    float tau = (aa + sqrtf(fmaxf(8.f - aa * aa, 0.f))) * 0.5f;
    tau = (tau <= aa) ? 2.f : tau;
    float t = fmaxf(tau - aa, 0.f);
    float yneg = 0.25f * t * t;
    return (xg >= 0.f) ? 1.f - yneg : yneg;
}

__global__ __launch_bounds__(256) void k_gemm_gate(const __bf16* __restrict__ Xbf,
                                                   const __bf16* __restrict__ FW,
                                                   const float* __restrict__ thr,
                                                   const float* __restrict__ ltemp,
                                                   float* __restrict__ gate_buf,
                                                   int finpad) {
    __shared__ __align__(16) __bf16 As[2][128][32];
    __shared__ __align__(16) __bf16 Bs[2][128][32];

    int tid  = threadIdx.x;
    int lane = tid & 31;
    int wid  = tid >> 5;
    int wm   = wid & 3;          // 0..3 -> 32-row slab
    int wn   = wid >> 2;         // 0..1 -> 64-col slab
    int bm   = blockIdx.x;
    int bn   = blockIdx.y;

    int lrow  = lane & 15;
    int khalf = (lane >> 4) * 8;

    int r    = tid >> 1;
    int koff = (tid & 1) * 16;
    const __bf16* aPtr = Xbf + (size_t)(bm * 128 + r) * KSTRIDE + koff;
    const __bf16* bPtr = FW  + (size_t)(bn * 128 + r) * finpad  + koff;

    f32x8 C[2][4] = {};

    const int ntiles = finpad / 32;

    // prologue: stage tile 0 into buffer 0
    {
        bf16x8 a0 = *(const bf16x8*)(aPtr);
        bf16x8 a1 = *(const bf16x8*)(aPtr + 8);
        bf16x8 b0 = *(const bf16x8*)(bPtr);
        bf16x8 b1 = *(const bf16x8*)(bPtr + 8);
        *(bf16x8*)&As[0][r][koff]     = a0;
        *(bf16x8*)&As[0][r][koff + 8] = a1;
        *(bf16x8*)&Bs[0][r][koff]     = b0;
        *(bf16x8*)&Bs[0][r][koff + 8] = b1;
    }

    for (int t = 0; t < ntiles; ++t) {
        __syncthreads();                       // buffer (t&1) now valid
        int cur = t & 1;
        int nxt = cur ^ 1;
        bool more = (t + 1 < ntiles);

        // issue next tile's global loads early (L2-resident; overlaps WMMAs)
        bf16x8 na0, na1, nb0, nb1;
        if (more) {
            int kg = (t + 1) * 32;
            na0 = *(const bf16x8*)(aPtr + kg);
            na1 = *(const bf16x8*)(aPtr + kg + 8);
            nb0 = *(const bf16x8*)(bPtr + kg);
            nb1 = *(const bf16x8*)(bPtr + kg + 8);
        }

        // fragment assembly per CDNA5 16-bit A/B layout:
        // lane<16 holds K in {0..7,16..23}, lane>=16 holds K in {8..15,24..31}
        bf16x16 afrag[2], bfrag[4];
        #pragma unroll
        for (int i = 0; i < 2; ++i) {
            int rr = wm * 32 + i * 16 + lrow;
            bf16x8 lo = *(const bf16x8*)&As[cur][rr][khalf];
            bf16x8 hi = *(const bf16x8*)&As[cur][rr][khalf + 16];
            #pragma unroll
            for (int e = 0; e < 8; ++e) { afrag[i][e] = lo[e]; afrag[i][e + 8] = hi[e]; }
        }
        #pragma unroll
        for (int j = 0; j < 4; ++j) {
            int cc = wn * 64 + j * 16 + lrow;
            bf16x8 lo = *(const bf16x8*)&Bs[cur][cc][khalf];
            bf16x8 hi = *(const bf16x8*)&Bs[cur][cc][khalf + 16];
            #pragma unroll
            for (int e = 0; e < 8; ++e) { bfrag[j][e] = lo[e]; bfrag[j][e + 8] = hi[e]; }
        }

        #pragma unroll
        for (int i = 0; i < 2; ++i)
            #pragma unroll
            for (int j = 0; j < 4; ++j)
                C[i][j] = __builtin_amdgcn_wmma_f32_16x16x32_bf16(
                    false, afrag[i], false, bfrag[j], (short)0, C[i][j], false, false);

        if (more) {
            *(bf16x8*)&As[nxt][r][koff]     = na0;
            *(bf16x8*)&As[nxt][r][koff + 8] = na1;
            *(bf16x8*)&Bs[nxt][r][koff]     = nb0;
            *(bf16x8*)&Bs[nxt][r][koff + 8] = nb1;
        }
    }

    // --- epilogue: entmoid gates straight from accumulators ---
    #pragma unroll
    for (int j = 0; j < 4; ++j) {
        int colg = bn * 128 + wn * 64 + j * 16 + lrow;
        float th = thr[colg];
        float sc = __expf(-ltemp[colg]);
        #pragma unroll
        for (int i = 0; i < 2; ++i) {
            int rbase = bm * 128 + wm * 32 + i * 16 + (lane >> 4) * 8;
            #pragma unroll
            for (int v = 0; v < 8; ++v) {
                float g = entmoid15_dev((C[i][j][v] - th) * sc);
                gate_buf[(size_t)(rbase + v) * NCOLS + colg] = g;
            }
        }
    }
}

// ---------------------------------------------------------------------------
// Leaves: per (b, tree) compute 64-leaf routing weights, dot with resp(64x3),
// write bf16 h into Xbf columns [colbase, colbase+192), reduce channel-0 and
// the gate regularizer.
// ---------------------------------------------------------------------------
__global__ __launch_bounds__(256) void k_leaves(const float* __restrict__ gate_buf,
                                                const float* __restrict__ resp,
                                                __bf16* __restrict__ Xbf,
                                                float* __restrict__ outv,
                                                float* __restrict__ regAcc,
                                                int colbase, int layer) {
    __shared__ float s0[256];
    __shared__ float sr[256];

    int tid = threadIdx.x;
    int n   = tid & 63;
    int bl  = tid >> 6;
    int b   = blockIdx.x * 4 + bl;

    const float* gp = gate_buf + (size_t)b * NCOLS + n * TDEPTH;
    float g[TDEPTH];
    float regs = 0.f;
    #pragma unroll
    for (int j = 0; j < TDEPTH; ++j) {
        g[j] = gp[j];
        regs += g[j] * (1.f - g[j]);
    }

    const float* rp = resp + n * (TREE_DIM * 64);
    float acc0 = 0.f, acc1 = 0.f, acc2 = 0.f;
    for (int l = 0; l < 64; ++l) {
        float w = 1.f;
        #pragma unroll
        for (int j = 0; j < TDEPTH; ++j)
            w *= ((l >> (5 - j)) & 1) ? (1.f - g[j]) : g[j];
        acc0 += w * rp[l];
        acc1 += w * rp[64 + l];
        acc2 += w * rp[128 + l];
    }

    __bf16* xr = Xbf + (size_t)b * KSTRIDE + colbase + n * TREE_DIM;
    xr[0] = (__bf16)acc0;
    xr[1] = (__bf16)acc1;
    xr[2] = (__bf16)acc2;

    s0[tid] = acc0;
    sr[tid] = regs;
    __syncthreads();
    for (int st = 32; st > 0; st >>= 1) {        // per-b (64-thread group) reduce
        if ((tid & 63) < st) s0[tid] += s0[tid + st];
        __syncthreads();
    }
    for (int st = 128; st > 0; st >>= 1) {       // whole-block reduce for reg
        if (tid < st) sr[tid] += sr[tid + st];
        __syncthreads();
    }
    if (n == 0) atomicAdd(&outv[b], s0[tid] * (1.f / (float)NCOLS));
    if (tid == 0) atomicAdd(&regAcc[layer], sr[0]);
}

// ---------------------------------------------------------------------------
__global__ void k_finish(const float* __restrict__ regAcc, float* __restrict__ outv) {
    if (threadIdx.x == 0 && blockIdx.x == 0) {
        float s = 0.f;
        for (int i = 0; i < NLAYERS; ++i) s += regAcc[i];
        outv[BATCH] = s / (float)((size_t)BATCH * NTREES * TDEPTH);
    }
}

// ---------------------------------------------------------------------------
extern "C" void kernel_launch(void* const* d_in, const int* in_sizes, int n_in,
                              void* d_out, int out_size, void* d_ws, size_t ws_size,
                              hipStream_t stream) {
    (void)in_sizes; (void)n_in; (void)out_size; (void)ws_size;
    const float* x = (const float*)d_in[0];
    float* outv = (float*)d_out;

    char* ws = (char*)d_ws;
    size_t o = 0;
    __bf16* Xbf = (__bf16*)(ws + o);
    o += (size_t)BATCH * KSTRIDE * sizeof(__bf16);      o = (o + 255) & ~(size_t)255;
    __bf16* FW = (__bf16*)(ws + o);
    o += (size_t)KSTRIDE * NCOLS * sizeof(__bf16);      o = (o + 255) & ~(size_t)255;
    float* gate_buf = (float*)(ws + o);
    o += (size_t)BATCH * NCOLS * sizeof(float);         o = (o + 255) & ~(size_t)255;
    float* regAcc = (float*)(ws + o);

    const int perB = 96 + (NLAYERS - 1) * 6;
    k_init<<<(BATCH * perB + 255) / 256, 256, 0, stream>>>(x, Xbf, outv, regAcc);

    int fin = 90;
    for (int i = 0; i < NLAYERS; ++i) {
        int finpad = (fin + 31) & ~31;
        const float* fsl   = (const float*)d_in[1 + 4 * i];
        const float* thr   = (const float*)d_in[2 + 4 * i];
        const float* ltemp = (const float*)d_in[3 + 4 * i];
        const float* resp  = (const float*)d_in[4 + 4 * i];

        k_entmax<<<NCOLS, 256, 0, stream>>>(fsl, FW, fin, finpad);

        dim3 grid(BATCH / 128, NCOLS / 128);
        k_gemm_gate<<<grid, 256, 0, stream>>>(Xbf, FW, thr, ltemp, gate_buf, finpad);

        k_leaves<<<BATCH / 4, 256, 0, stream>>>(gate_buf, resp, Xbf, outv, regAcc, fin, i);

        fin += NTREES * TREE_DIM;
    }
    k_finish<<<1, 32, 0, stream>>>(regAcc, outv);
}
